// MolTransformer_88347477279423
// MI455X (gfx1250) — matrix operand order repaired and checked by reading, not compiled, not run on hardware
//
#include <hip/hip_runtime.h>
#include <hip/hip_bf16.h>

// MolTransformer for gfx1250 (MI455X). GEMMs use V_WMMA_F32_16X16X32_F16
// with f32 accumulate. A-tiles are converted to f16 once at staging time and
// stored in LDS in WMMA-fragment order (2x ds_load_b128 per fragment).
// Persistent blocks keep the 64x64 weight matrix resident in VGPRs and
// software-pipeline the next A-tile's global loads against WMMA compute.

typedef __attribute__((ext_vector_type(16))) _Float16 v16h;
typedef __attribute__((ext_vector_type(8)))  _Float16 v8h;
typedef __attribute__((ext_vector_type(4)))  _Float16 v4h;
typedef __attribute__((ext_vector_type(8)))  float    v8f;

#define DIMD   64
#define SEQ    64
#define HEADS  8
#define DH     8
#define LAYERS 4
#define LDSW   68   // f32 LDS row stride (floats) to avoid bank conflicts
#define FRAGP  24   // halves per fragment slot (16 used, padded for banks)

// ---------------------------------------------------------------- utilities

__global__ void k_counts(const unsigned char* __restrict__ mask,
                         int* __restrict__ counts, int B) {
    int b = blockIdx.x * blockDim.x + threadIdx.x;
    if (b < B) {
        int c = 0;
        #pragma unroll
        for (int s = 0; s < SEQ; ++s) c += (int)mask[b * SEQ + s];
        counts[b] = c;
    }
}

__global__ void k_offsets(const int* __restrict__ counts,
                          int* __restrict__ offsets, int B) {
    if (blockIdx.x == 0 && threadIdx.x == 0) {
        int acc = 0;
        for (int b = 0; b < B; ++b) { offsets[b] = acc; acc += counts[b]; }
    }
}

// x[i,:] = emb[node_types[i],:]; deg[i] = 1 (self loop)
__global__ void k_embed(const int* __restrict__ nt, const float* __restrict__ emb,
                        float* __restrict__ x, float* __restrict__ deg, int N) {
    int gid = blockIdx.x * blockDim.x + threadIdx.x;
    int i = gid >> 6, c = gid & 63;
    if (i < N) {
        x[(size_t)i * DIMD + c] = emb[nt[i] * DIMD + c];
        if (c == 0) deg[i] = 1.0f;
    }
}

__global__ void k_deg(const int* __restrict__ dst, float* __restrict__ deg, int E) {
    int e = blockIdx.x * blockDim.x + threadIdx.x;
    if (e < E) atomicAdd(&deg[dst[e]], 1.0f);
}

__global__ void k_dinv(const float* __restrict__ deg, float* __restrict__ dinv, int N) {
    int i = blockIdx.x * blockDim.x + threadIdx.x;
    if (i < N) dinv[i] = rsqrtf(deg[i]);
}

// agg[i,:] = dinv[i]^2 * t[i,:]   (self-loop contribution)
__global__ void k_agg_init(const float* __restrict__ t, const float* __restrict__ dinv,
                           float* __restrict__ agg, int N) {
    int gid = blockIdx.x * blockDim.x + threadIdx.x;
    int i = gid >> 6, c = gid & 63;
    if (i < N) {
        float w = dinv[i];
        agg[(size_t)i * DIMD + c] = w * w * t[(size_t)i * DIMD + c];
    }
}

// agg[d,:] += dinv[s]*dinv[d] * t[s,:]
__global__ void k_agg_edges(const int* __restrict__ src, const int* __restrict__ dst,
                            const float* __restrict__ t, const float* __restrict__ dinv,
                            float* __restrict__ agg, int E) {
    int gid = blockIdx.x * blockDim.x + threadIdx.x;
    int e = gid >> 6, c = gid & 63;
    if (e < E) {
        int s = src[e], d = dst[e];
        float w = dinv[s] * dinv[d];
        atomicAdd(&agg[(size_t)d * DIMD + c], w * t[(size_t)s * DIMD + c]);
    }
}

__global__ void k_bias_act(const float* __restrict__ agg, const float* __restrict__ b,
                           float* __restrict__ out, int N, int relu) {
    int gid = blockIdx.x * blockDim.x + threadIdx.x;
    int i = gid >> 6, c = gid & 63;
    if (i < N) {
        float v = agg[(size_t)i * DIMD + c] + b[c];
        if (relu) v = fmaxf(v, 0.0f);
        out[(size_t)i * DIMD + c] = v;
    }
}

// h[b,s,:] = mask[b,s] ? x[offsets[b]+s,:] : 0   (nodes contiguous per graph)
__global__ void k_pack(const float* __restrict__ x, const unsigned char* __restrict__ mask,
                       const int* __restrict__ offsets, float* __restrict__ h, int R) {
    int gid = blockIdx.x * blockDim.x + threadIdx.x;
    int row = gid >> 6, c = gid & 63;   // row = b*SEQ + s
    if (row < R) {
        int b = row >> 6, s = row & 63;
        float v = 0.0f;
        if (mask[row]) v = x[(size_t)(offsets[b] + s) * DIMD + c];
        h[(size_t)row * DIMD + c] = v;
    }
}

// ------------------------------------------------------------ WMMA GEMM 64
// C[M x 64] = A[M x 64] @ W[64 x 64] (+bias)(+relu), f32 in/out, f16 WMMA.
// Persistent blocks: 128 thr = 4 waves; B fragments loaded once per block;
// grid-stride over 64-row tiles with register prefetch of the next tile.
// M == numTiles*64 exactly (caller pads). FLAGS: bit0 = bias, bit1 = relu.
template <int FLAGS>
__global__ void __launch_bounds__(128)
k_gemm64(const float* __restrict__ A, const float* __restrict__ W,
         const float* __restrict__ bias, float* __restrict__ C, int numTiles) {
    // A staged as f16 in WMMA-fragment order: slot = (rt*2 + frag)*32 + lane,
    // 16 halves per slot, padded to FRAGP for bank spread (48B stride, 16B aligned)
    __shared__ __align__(16) _Float16 Af[8 * 32 * FRAGP];
    __shared__ float Cs[64 * LDSW];

    const int tid  = threadIdx.x;
    const int lane = tid & 31;
    const int half = lane >> 4;
    const int nn   = lane & 15;
    const int n0   = (tid >> 5) * 16;

    // ---- B fragments: loaded & converted ONCE per persistent block.
    // Layout: lane nn owns column n0+nn; lanes>=16 hold K upper half (16..31).
    v16h b0, b1;
    #pragma unroll
    for (int j = 0; j < 16; ++j) {
        int kA = half * 16 + j;
        b0[j] = (_Float16)W[(kA +  0) * DIMD + n0 + nn];
        b1[j] = (_Float16)W[(kA + 32) * DIMD + n0 + nn];
    }
    float bv = 0.0f;
    if (FLAGS & 1) bv = bias[n0 + nn];
    const int colg = n0 + nn;

    // Per-thread staging geometry (8 float4 chunks covering the 64x64 tile)
    int srow[8], scol[8], sslot[8], sj[8];
    #pragma unroll
    for (int it = 0; it < 8; ++it) {
        int idx = (it * 128 + tid) * 4;
        int r = idx >> 6, c = idx & 63;
        srow[it] = r; scol[it] = c;
        int f  = c >> 5;            // which K=32 fragment
        int kk = c & 31;
        int hf = kk >> 4;           // lane half that owns this k-group
        sj[it]    = kk & 15;
        sslot[it] = (((r >> 4) << 1) + f) * 32 + ((r & 15) + (hf << 4));
    }

    const int G = gridDim.x;
    float4 regs[8];

    // Prefetch first tile (grid is sized <= numTiles, so always valid)
    {
        size_t row0 = (size_t)blockIdx.x * 64;
        #pragma unroll
        for (int it = 0; it < 8; ++it)
            regs[it] = *reinterpret_cast<const float4*>(A + (row0 + srow[it]) * DIMD + scol[it]);
    }

    for (int t = blockIdx.x; t < numTiles; t += G) {
        // ---- stage current tile: convert f32->f16 into fragment order
        #pragma unroll
        for (int it = 0; it < 8; ++it) {
            v4h h4;
            h4[0] = (_Float16)regs[it].x;
            h4[1] = (_Float16)regs[it].y;
            h4[2] = (_Float16)regs[it].z;
            h4[3] = (_Float16)regs[it].w;
            *reinterpret_cast<v4h*>(&Af[sslot[it] * FRAGP + sj[it]]) = h4;
        }
        __syncthreads();

        // ---- issue next tile's global loads (waited on at next stage step)
        int tn = t + G;
        if (tn < numTiles) {
            size_t row0n = (size_t)tn * 64;
            #pragma unroll
            for (int it = 0; it < 8; ++it)
                regs[it] = *reinterpret_cast<const float4*>(A + (row0n + srow[it]) * DIMD + scol[it]);
        }

        // ---- compute 4 row tiles; all waves share the same A fragments
        #pragma unroll
        for (int rt = 0; rt < 4; ++rt) {
            const int s0 = ((rt << 1) + 0) * 32 + lane;
            const int s1 = ((rt << 1) + 1) * 32 + lane;
            v8h lo0 = *reinterpret_cast<const v8h*>(&Af[s0 * FRAGP + 0]);
            v8h hi0 = *reinterpret_cast<const v8h*>(&Af[s0 * FRAGP + 8]);
            v8h lo1 = *reinterpret_cast<const v8h*>(&Af[s1 * FRAGP + 0]);
            v8h hi1 = *reinterpret_cast<const v8h*>(&Af[s1 * FRAGP + 8]);
            v16h a0, a1;
            #pragma unroll
            for (int j = 0; j < 8; ++j) {
                a0[j] = lo0[j]; a0[j + 8] = hi0[j];
                a1[j] = lo1[j]; a1[j + 8] = hi1[j];
            }
            v8f c = {};
            c = __builtin_amdgcn_wmma_f32_16x16x32_f16(false, a0, false, b0, (short)0, c, false, false);
            c = __builtin_amdgcn_wmma_f32_16x16x32_f16(false, a1, false, b1, (short)0, c, false, false);

            // C/D layout: lane<16 -> (M=r, N=lane); lane>=16 -> (M=r+8, N=lane-16)
            #pragma unroll
            for (int r = 0; r < 8; ++r) {
                float v = c[r] + bv;
                if (FLAGS & 2) v = fmaxf(v, 0.0f);
                Cs[(rt * 16 + half * 8 + r) * LDSW + colg] = v;
            }
        }
        __syncthreads();

        // ---- coalesced b128 stores of the transposed-in-LDS C tile
        {
            size_t row0 = (size_t)t * 64;
            #pragma unroll
            for (int it = 0; it < 8; ++it) {
                int r = srow[it], c = scol[it];
                float4 v = make_float4(Cs[r * LDSW + c + 0], Cs[r * LDSW + c + 1],
                                       Cs[r * LDSW + c + 2], Cs[r * LDSW + c + 3]);
                *reinterpret_cast<float4*>(C + (row0 + r) * DIMD + c) = v;
            }
        }
        __syncthreads();   // Af/Cs may be overwritten next iteration
    }
}

// ------------------------------------------------------------ attention
// One block per (batch, head): 64 threads, thread t = query row.
__global__ void __launch_bounds__(64)
k_attn(const float* __restrict__ q, const float* __restrict__ kbuf,
       const float* __restrict__ vbuf, const unsigned char* __restrict__ mask,
       float* __restrict__ o) {
    __shared__ float ks[SEQ][DH];
    __shared__ float vs[SEQ][DH];
    __shared__ float sc[SEQ][SEQ + 1];
    __shared__ float mk[SEQ];
    const int b  = blockIdx.x >> 3;
    const int hh = blockIdx.x & 7;
    const int t  = threadIdx.x;
    const int rowBase = b * SEQ;

    {
        const float* kr = kbuf + (size_t)(rowBase + t) * DIMD + hh * DH;
        const float* vr = vbuf + (size_t)(rowBase + t) * DIMD + hh * DH;
        #pragma unroll
        for (int d = 0; d < DH; ++d) { ks[t][d] = kr[d]; vs[t][d] = vr[d]; }
        mk[t] = mask[rowBase + t] ? 1.0f : 0.0f;
    }
    __syncthreads();

    float qv[DH];
    const float* qr = q + (size_t)(rowBase + t) * DIMD + hh * DH;
    #pragma unroll
    for (int d = 0; d < DH; ++d) qv[d] = qr[d];

    const float scale = 0.3535533905932738f;  // 1/sqrt(8)
    float mx = -3.0e38f;
    for (int j = 0; j < SEQ; ++j) {
        float s = 0.0f;
        #pragma unroll
        for (int d = 0; d < DH; ++d) s += qv[d] * ks[j][d];
        s = (mk[j] > 0.5f) ? s * scale : -1.0e9f;
        sc[t][j] = s;
        mx = fmaxf(mx, s);
    }
    float sum = 0.0f;
    float acc[DH] = {};
    for (int j = 0; j < SEQ; ++j) {
        float e = expf(sc[t][j] - mx);
        sum += e;
        #pragma unroll
        for (int d = 0; d < DH; ++d) acc[d] += e * vs[j][d];
    }
    float inv = 1.0f / sum;
    float* orow = o + (size_t)(rowBase + t) * DIMD + hh * DH;
    #pragma unroll
    for (int d = 0; d < DH; ++d) orow[d] = acc[d] * inv;
}

// ---------------------------------------------------- residual + layernorm
// h[row] = LN(h[row] + r[row]) * g + b ; one wave32 per row, 2 elems/lane.
__global__ void __launch_bounds__(256)
k_add_ln(float* __restrict__ h, const float* __restrict__ r,
         const float* __restrict__ g, const float* __restrict__ bta, int R) {
    const int wave = threadIdx.x >> 5;
    const int lane = threadIdx.x & 31;
    const int row  = blockIdx.x * 8 + wave;
    if (row >= R) return;
    size_t base = (size_t)row * DIMD;
    float e0 = h[base + lane]      + r[base + lane];
    float e1 = h[base + lane + 32] + r[base + lane + 32];
    float s  = e0 + e1;
    float s2 = e0 * e0 + e1 * e1;
    #pragma unroll
    for (int off = 16; off >= 1; off >>= 1) {
        s  += __shfl_xor(s,  off, 32);
        s2 += __shfl_xor(s2, off, 32);
    }
    float mu  = s  * (1.0f / 64.0f);
    float var = s2 * (1.0f / 64.0f) - mu * mu;
    float rs  = rsqrtf(var + 1e-5f);
    h[base + lane]      = (e0 - mu) * rs * g[lane]      + bta[lane];
    h[base + lane + 32] = (e1 - mu) * rs * g[lane + 32] + bta[lane + 32];
}

// -------------------------------------------------------------------- host

static inline int gemmGrid(int numTiles) {
    return numTiles < 1024 ? numTiles : 1024;
}

extern "C" void kernel_launch(void* const* d_in, const int* in_sizes, int n_in,
                              void* d_out, int out_size, void* d_ws, size_t ws_size,
                              hipStream_t stream) {
    const int*           node_types = (const int*)d_in[0];
    const int*           edge_index = (const int*)d_in[1];
    const unsigned char* node_mask  = (const unsigned char*)d_in[2];
    const float* emb  = (const float*)d_in[3];
    const float* gW1  = (const float*)d_in[4];
    const float* gb1  = (const float*)d_in[5];
    const float* gW2  = (const float*)d_in[6];
    const float* gb2  = (const float*)d_in[7];
    const float* Wq   = (const float*)d_in[8];
    const float* bq   = (const float*)d_in[9];
    const float* Wk   = (const float*)d_in[10];
    const float* bk   = (const float*)d_in[11];
    const float* Wv   = (const float*)d_in[12];
    const float* bv   = (const float*)d_in[13];
    const float* Wo   = (const float*)d_in[14];
    const float* bo   = (const float*)d_in[15];
    const float* Wf1  = (const float*)d_in[16];
    const float* bf1  = (const float*)d_in[17];
    const float* Wf2  = (const float*)d_in[18];
    const float* bf2  = (const float*)d_in[19];
    const float* ln1g = (const float*)d_in[20];
    const float* ln1b = (const float*)d_in[21];
    const float* ln2g = (const float*)d_in[22];
    const float* ln2b = (const float*)d_in[23];

    const int N = in_sizes[0];          // nodes
    const int E = in_sizes[1] / 2;      // edges
    const int R = in_sizes[2];          // B*S dense rows
    const int B = R / SEQ;
    const int* esrc = edge_index;
    const int* edst = edge_index + E;
    (void)n_in; (void)out_size; (void)ws_size;

    float* h = (float*)d_out;           // dense activations live in d_out
    char* ws = (char*)d_ws;
    size_t bufBytes = (size_t)R * DIMD * sizeof(float);
    float* qb   = (float*)(ws);
    float* kb   = (float*)(ws + bufBytes);
    float* vb   = (float*)(ws + 2 * bufBytes);
    float* ob   = (float*)(ws + 3 * bufBytes);
    float* deg  = (float*)(ws + 4 * bufBytes);
    float* dinv = deg + R;
    int* counts  = (int*)(dinv + R);
    int* offsets = counts + B;

    // ---- graph preprocessing
    k_counts <<<(B + 255) / 256, 256, 0, stream>>>(node_mask, counts, B);
    k_offsets<<<1, 1, 0, stream>>>(counts, offsets, B);
    k_embed  <<<(N * 64 + 255) / 256, 256, 0, stream>>>(node_types, emb, qb, deg, N);
    k_deg    <<<(E + 255) / 256, 256, 0, stream>>>(edst, deg, E);
    k_dinv   <<<(N + 255) / 256, 256, 0, stream>>>(deg, dinv, N);

    const int tilesN = (N + 63) / 64;   // GEMM runs on padded rows (in-bounds of R-row buffers)
    // ---- GCN layer 1 (relu applied after aggregation)
    k_gemm64<0><<<gemmGrid(tilesN), 128, 0, stream>>>(qb, gW1, nullptr, kb, tilesN);
    k_agg_init <<<(N * 64 + 255) / 256, 256, 0, stream>>>(kb, dinv, vb, N);
    k_agg_edges<<<(E * 64 + 255) / 256, 256, 0, stream>>>(esrc, edst, kb, dinv, vb, E);
    k_bias_act <<<(N * 64 + 255) / 256, 256, 0, stream>>>(vb, gb1, qb, N, 1);
    // ---- GCN layer 2
    k_gemm64<0><<<gemmGrid(tilesN), 128, 0, stream>>>(qb, gW2, nullptr, kb, tilesN);
    k_agg_init <<<(N * 64 + 255) / 256, 256, 0, stream>>>(kb, dinv, vb, N);
    k_agg_edges<<<(E * 64 + 255) / 256, 256, 0, stream>>>(esrc, edst, kb, dinv, vb, E);
    k_bias_act <<<(N * 64 + 255) / 256, 256, 0, stream>>>(vb, gb2, qb, N, 0);
    // ---- pack to dense
    k_pack<<<(R * 64 + 255) / 256, 256, 0, stream>>>(qb, node_mask, offsets, h, R);

    const int tilesR = R / 64;
    const int gR = gemmGrid(tilesR);
    for (int l = 0; l < LAYERS; ++l) {
        const float* Wq_l = Wq + l * DIMD * DIMD;  const float* bq_l = bq + l * DIMD;
        const float* Wk_l = Wk + l * DIMD * DIMD;  const float* bk_l = bk + l * DIMD;
        const float* Wv_l = Wv + l * DIMD * DIMD;  const float* bv_l = bv + l * DIMD;
        const float* Wo_l = Wo + l * DIMD * DIMD;  const float* bo_l = bo + l * DIMD;
        const float* W1_l = Wf1 + l * DIMD * DIMD; const float* b1_l = bf1 + l * DIMD;
        const float* W2_l = Wf2 + l * DIMD * DIMD; const float* b2_l = bf2 + l * DIMD;

        k_gemm64<1><<<gR, 128, 0, stream>>>(h, Wq_l, bq_l, qb, tilesR);
        k_gemm64<1><<<gR, 128, 0, stream>>>(h, Wk_l, bk_l, kb, tilesR);
        k_gemm64<1><<<gR, 128, 0, stream>>>(h, Wv_l, bv_l, vb, tilesR);
        k_attn  <<<B * HEADS, SEQ, 0, stream>>>(qb, kb, vb, node_mask, ob);
        k_gemm64<1><<<gR, 128, 0, stream>>>(ob, Wo_l, bo_l, qb, tilesR);
        k_add_ln<<<R / 8, 256, 0, stream>>>(h, qb, ln1g + l * DIMD, ln1b + l * DIMD, R);
        k_gemm64<3><<<gR, 128, 0, stream>>>(h, W1_l, b1_l, kb, tilesR);   // bias+relu
        k_gemm64<1><<<gR, 128, 0, stream>>>(kb, W2_l, b2_l, vb, tilesR);
        k_add_ln<<<R / 8, 256, 0, stream>>>(h, vb, ln2g + l * DIMD, ln2b + l * DIMD, R);
    }
}
